// MixtureOfAggregators_12910671692273
// MI455X (gfx1250) — matrix operand order repaired and checked by reading, not compiled, not run on hardware
//
#include <hip/hip_runtime.h>

// ---------------------------------------------------------------------------
// MixtureOfAggregators on MI455X (gfx1250): bf16 WMMA pipeline with
// double-buffered async global->LDS staging in the GEMM.
// E=4 experts, L=2 layers, H=8 heads, DH=64, D=512, DIN=2048, MLP=512,
// B=4 bags, N=1024 patches, S=N+1=1025 tokens. NC=4 classes.
// ---------------------------------------------------------------------------

typedef __bf16 bf16;
typedef __attribute__((ext_vector_type(16))) __bf16 v16bf;
typedef __attribute__((ext_vector_type(8)))  __bf16 v8bf;
typedef __attribute__((ext_vector_type(8)))  float  v8f;

#define ACT_NONE 0
#define ACT_RELU 1
#define ACT_GELU 2

static __device__ __forceinline__ v8f wmma_bf16(v16bf a, v16bf b, v8f c) {
  return __builtin_amdgcn_wmma_f32_16x16x32_bf16(false, a, false, b, (short)0, c,
                                                 false, false);
}

static __device__ __forceinline__ v16bf ld16c(const bf16* p) {
  // 16 contiguous bf16 (32B) -> two b128 loads
  v8bf lo = *(const v8bf*)p;
  v8bf hi = *(const v8bf*)(p + 8);
  return __builtin_shufflevector(lo, hi, 0,1,2,3,4,5,6,7,8,9,10,11,12,13,14,15);
}
static __device__ __forceinline__ v16bf ld8x2(const bf16* p, int gap_halves) {
  v8bf lo = *(const v8bf*)p;
  v8bf hi = *(const v8bf*)(p + gap_halves);
  return __builtin_shufflevector(lo, hi, 0,1,2,3,4,5,6,7,8,9,10,11,12,13,14,15);
}

// Async global->LDS 16B copy (per lane), tracked by ASYNCcnt.
static __device__ __forceinline__ void async_b128(unsigned lds_off,
                                                  const bf16* gptr) {
  asm volatile("global_load_async_to_lds_b128 %0, %1, off"
               :
               : "v"(lds_off), "v"((unsigned long long)(uintptr_t)gptr)
               : "memory");
}
static __device__ __forceinline__ void wait_async0() {
  asm volatile("s_wait_asynccnt 0x0" ::: "memory");
}

static __device__ __forceinline__ float wave_sum(float v) {
  v += __shfl_xor(v, 1, 32);
  v += __shfl_xor(v, 2, 32);
  v += __shfl_xor(v, 4, 32);
  v += __shfl_xor(v, 8, 32);
  v += __shfl_xor(v, 16, 32);
  return v;
}

static __device__ __forceinline__ float gelu_tanh(float x) {
  float x3 = x * x * x;
  return 0.5f * x * (1.f + tanhf(0.7978845608028654f * (x + 0.044715f * x3)));
}

// ---------------------------------------------------------------------------
// fp32 -> bf16 conversion (grid-stride)
// ---------------------------------------------------------------------------
__global__ __launch_bounds__(256) void cvt_kernel(const float* __restrict__ in,
                                                  bf16* __restrict__ out, long n) {
  long i = (long)blockIdx.x * 256 + threadIdx.x;
  long stride = (long)gridDim.x * 256;
  for (; i < n; i += stride) out[i] = (bf16)in[i];
}

// ---------------------------------------------------------------------------
// Tiled bf16 GEMM: C[M,N] = act(A[M,K] @ B[K,N] + bias[N]) (+ residual)
// Block tile 128x64, BK=64, 8 waves each 32x32 (2x2 WMMA frags).
// Double-buffered async global->LDS staging (ASYNCcnt pipeline).
// REMAP: output row m -> m + m/remapN + 1 (patch rows into [B,S,D] past CLS).
// ---------------------------------------------------------------------------
template <int ACT, bool RESID, bool OUT_BF16, bool REMAP>
__global__ __launch_bounds__(256) void gemm_kernel(
    const bf16* __restrict__ A, const bf16* __restrict__ B,
    const float* __restrict__ bias, float* __restrict__ Cf,
    bf16* __restrict__ Cb, const float* __restrict__ resid,
    int M, int N, int K, int remapN) {
  __shared__ bf16 As[2][128 * 72];  // 64 halves per row, padded to 72
  __shared__ bf16 Bs[2][64 * 72];

  const int tid  = threadIdx.x;
  const int lane = tid & 31;
  const int wave = tid >> 5;
  const int wm = wave >> 1;   // 0..3
  const int wn = wave & 1;    // 0..1
  const int mi = lane & 15;
  const int hs = lane >> 4;   // lane half-select

  const int m0 = blockIdx.y * 128;
  const int n0 = blockIdx.x * 64;

  v8f acc[2][2];
#pragma unroll
  for (int f = 0; f < 2; f++)
#pragma unroll
    for (int g = 0; g < 2; g++)
#pragma unroll
      for (int i = 0; i < 8; i++) acc[f][g][i] = 0.f;

  const int ar  = tid >> 1;         // 0..127 (A tile row)
  const int ac  = (tid & 1) * 32;   // halves
  const int brr = tid >> 2;         // 0..63 (B tile row)
  const int bcc = (tid & 3) * 16;   // halves

  // OOB M-rows: clamp source row (garbage only affects unstored C rows).
  const int arow = min(m0 + ar, M - 1);
  const bf16* gA = A + (size_t)arow * K + ac;
  const bf16* gB = B + (size_t)brr * N + n0 + bcc;

  const unsigned ldsA = (unsigned)(uintptr_t)(&As[0][0]) + (ar * 72 + ac) * 2;
  const unsigned ldsB = (unsigned)(uintptr_t)(&Bs[0][0]) + (brr * 72 + bcc) * 2;
  const unsigned bufA = 128 * 72 * 2;  // bytes per A buffer
  const unsigned bufB = 64 * 72 * 2;   // bytes per B buffer

  const int nk = K >> 6;

  // prologue: stage tile 0 into buffer 0
  {
#pragma unroll
    for (int s = 0; s < 4; s++) async_b128(ldsA + s * 16, gA + s * 8);
#pragma unroll
    for (int s = 0; s < 2; s++) async_b128(ldsB + s * 16, gB + s * 8);
  }

  for (int t = 0; t < nk; t++) {
    const int buf = t & 1;
    wait_async0();      // own async loads of tile t complete
    __syncthreads();    // everyone's tile t visible; prev buffer free

    if (t + 1 < nk) {
      const int k1 = (t + 1) << 6;
      const unsigned bsel = ((t + 1) & 1);
#pragma unroll
      for (int s = 0; s < 4; s++)
        async_b128(ldsA + bsel * bufA + s * 16, gA + k1 + s * 8);
#pragma unroll
      for (int s = 0; s < 2; s++)
        async_b128(ldsB + bsel * bufB + s * 16, gB + (size_t)k1 * N + s * 8);
    }

#pragma unroll
    for (int ks = 0; ks < 64; ks += 32) {
      v16bf afr[2], bfr[2];
#pragma unroll
      for (int f = 0; f < 2; f++) {
        const bf16* ap = &As[buf][(wm * 32 + f * 16 + mi) * 72 + ks + 8 * hs];
        afr[f] = ld8x2(ap, 16);
      }
#pragma unroll
      for (int g = 0; g < 2; g++) {
#pragma unroll
        for (int s = 0; s < 16; s++)
          bfr[g][s] = Bs[buf][(ks + (s & 7) + 8 * hs + ((s >> 3) << 4)) * 72 + wn * 32 + g * 16 + mi];
      }
#pragma unroll
      for (int f = 0; f < 2; f++)
#pragma unroll
        for (int g = 0; g < 2; g++) acc[f][g] = wmma_bf16(afr[f], bfr[g], acc[f][g]);
    }
  }

  // --- epilogue ---
#pragma unroll
  for (int f = 0; f < 2; f++) {
#pragma unroll
    for (int g = 0; g < 2; g++) {
      const int col = n0 + wn * 32 + g * 16 + mi;
      const float bv = bias[col];
#pragma unroll
      for (int i = 0; i < 8; i++) {
        const int gr = m0 + wm * 32 + f * 16 + i + 8 * hs;
        if (gr < M) {
          long orow = REMAP ? (long)gr + gr / remapN + 1 : (long)gr;
          float v = acc[f][g][i] + bv;
          if (RESID) v += resid[orow * N + col];
          if (ACT == ACT_RELU) v = fmaxf(v, 0.f);
          if (ACT == ACT_GELU) v = gelu_tanh(v);
          if (OUT_BF16) Cb[orow * N + col] = (bf16)v;
          else          Cf[orow * N + col] = v;
        }
      }
    }
  }
}

// ---------------------------------------------------------------------------
// LayerNorm over D=512, fp32 in -> bf16 out, one wave per row
// ---------------------------------------------------------------------------
__global__ __launch_bounds__(256) void ln_kernel(const float* __restrict__ h,
                                                 const float* __restrict__ g,
                                                 const float* __restrict__ b,
                                                 bf16* __restrict__ out, int rows) {
  const int wv = threadIdx.x >> 5, lane = threadIdx.x & 31;
  const int row = blockIdx.x * 8 + wv;
  if (row >= rows) return;
  const float* x = h + (size_t)row * 512;
  float s = 0.f, sq = 0.f;
#pragma unroll
  for (int j = 0; j < 16; j++) {
    float v = x[lane + j * 32];
    s += v; sq += v * v;
  }
  s = wave_sum(s); sq = wave_sum(sq);
  const float mean = s * (1.f / 512.f);
  const float var  = sq * (1.f / 512.f) - mean * mean;
  const float rs   = rsqrtf(var + 1e-5f);
  bf16* o = out + (size_t)row * 512;
#pragma unroll
  for (int j = 0; j < 16; j++) {
    int d = lane + j * 32;
    o[d] = (bf16)((x[d] - mean) * rs * g[d] + b[d]);
  }
}

// ---------------------------------------------------------------------------
// CLS row fill: h[b, 0, :] = cls[:]
// ---------------------------------------------------------------------------
__global__ __launch_bounds__(256) void clsfill_kernel(float* __restrict__ h,
                                                      const float* __restrict__ cls) {
  int i = blockIdx.x * 256 + threadIdx.x;
  if (i < 4 * 512) {
    int b = i >> 9, d = i & 511;
    h[(size_t)b * 1025 * 512 + d] = cls[d];
  }
}

// ---------------------------------------------------------------------------
// K transpose: kT[b,h,dh,s] = qkv[b,s, 512 + h*64 + dh], zero padded to Spad
// ---------------------------------------------------------------------------
__global__ __launch_bounds__(256) void kt_kernel(const bf16* __restrict__ qkv,
                                                 bf16* __restrict__ kT,
                                                 int S, int Spad) {
  int i = blockIdx.x * 256 + threadIdx.x;
  int total = 4 * 8 * 64 * Spad;
  if (i >= total) return;
  int s = i % Spad;
  int t = i / Spad;
  int dh = t & 63; t >>= 6;
  int hh = t & 7;
  int b = t >> 3;
  bf16 v = (bf16)0.f;
  if (s < S) v = qkv[(size_t)(b * S + s) * 1536 + 512 + hh * 64 + dh];
  kT[i] = v;
}

// ---------------------------------------------------------------------------
// Flash attention: one wave per (bag, head, 16-query tile). S=1025, Spad=1056.
// Scores = Q @ K^T via WMMA; online softmax via per-wave LDS tile; O += P @ V.
// Output: attn_out[b*S+s, h*64+dh] bf16.
// ---------------------------------------------------------------------------
__global__ __launch_bounds__(256) void attn_kernel(const bf16* __restrict__ qkv,
                                                   const bf16* __restrict__ kT,
                                                   bf16* __restrict__ attn_out,
                                                   int S, int Spad) {
  __shared__ float Stile[8][512];  // per-wave 16q x 32k scores (f32)
  __shared__ bf16  Ptile[8][512];  // per-wave 16q x 32k probs  (bf16)
  __shared__ float Arow[8][16];    // per-row rescale factor
  __shared__ float Lrow[8][16];    // per-row 1/l

  const int wv = threadIdx.x >> 5;
  const int lane = threadIdx.x & 31;
  const int QT = (S + 15) / 16;  // 65
  const int w = blockIdx.x * 8 + wv;
  if (w >= 4 * 8 * QT) return;

  const int qt = w % QT;
  const int hh = (w / QT) % 8;
  const int b  = w / (QT * 8);

  const int mi = lane & 15;
  const int hs = lane >> 4;
  const int q0 = qt * 16;

  // Q A-fragments (16 x 64, two K=32 substeps), clamped rows for tail tile
  const int qr_frag = min(q0 + mi, S - 1);
  const bf16* qbase = qkv + (size_t)(b * S + qr_frag) * 1536 + hh * 64;
  v16bf aq[2];
#pragma unroll
  for (int ks = 0; ks < 2; ks++) aq[ks] = ld8x2(qbase + ks * 32 + 8 * hs, 16);

  v8f accO[4];
#pragma unroll
  for (int g = 0; g < 4; g++)
#pragma unroll
    for (int i = 0; i < 8; i++) accO[g][i] = 0.f;

  float mrow = -1e30f, lrow = 0.f;  // owned by lanes < 16 (row = mi)
  const size_t ktbase = (size_t)(b * 8 + hh) * 64 * Spad;
  const float scale = 0.125f;  // 1/sqrt(64)

  for (int k0 = 0; k0 < Spad; k0 += 32) {
    // ---- scores: two 16x16 tiles ----
#pragma unroll
    for (int t = 0; t < 2; t++) {
      v8f sc;
#pragma unroll
      for (int i = 0; i < 8; i++) sc[i] = 0.f;
#pragma unroll
      for (int ks = 0; ks < 2; ks++) {
        v16bf kb;
#pragma unroll
        for (int s = 0; s < 16; s++)
          kb[s] = kT[ktbase + (size_t)(ks * 32 + (s & 7) + 8 * hs + ((s >> 3) << 4)) * Spad + k0 + t * 16 + mi];
        sc = wmma_bf16(aq[ks], kb, sc);
      }
#pragma unroll
      for (int i = 0; i < 8; i++)
        Stile[wv][(i + 8 * hs) * 32 + t * 16 + mi] = sc[i];
    }

    // ---- online softmax (lanes 0..15, one query row each) ----
    if (lane < 16) {
      float sv[32];
      float mx = mrow;
#pragma unroll
      for (int j = 0; j < 32; j++) {
        float v = Stile[wv][mi * 32 + j] * scale;
        if (k0 + j >= S) v = -1e30f;
        sv[j] = v;
        mx = fmaxf(mx, v);
      }
      float alpha = __expf(mrow - mx);
      float sum = 0.f;
#pragma unroll
      for (int j = 0; j < 32; j++) {
        float p = __expf(sv[j] - mx);
        sum += p;
        Ptile[wv][mi * 32 + j] = (bf16)p;
      }
      lrow = lrow * alpha + sum;
      mrow = mx;
      Arow[wv][mi] = alpha;
    }

    // ---- rescale accumulators ----
    float al[8];
#pragma unroll
    for (int i = 0; i < 8; i++) al[i] = Arow[wv][i + 8 * hs];
#pragma unroll
    for (int g = 0; g < 4; g++)
#pragma unroll
      for (int i = 0; i < 8; i++) accO[g][i] *= al[i];

    // ---- O += P @ V ----
    v16bf ap = ld8x2(&Ptile[wv][mi * 32 + 8 * hs], 16);
#pragma unroll
    for (int g = 0; g < 4; g++) {
      v16bf vb;
#pragma unroll
      for (int s = 0; s < 16; s++) {
        const int skc = min(k0 + (s & 7) + 8 * hs + ((s >> 3) << 4), S - 1);
        vb[s] = qkv[(size_t)(b * S + skc) * 1536 + 1024 + hh * 64 + g * 16 + mi];
      }
      accO[g] = wmma_bf16(ap, vb, accO[g]);
    }
  }

  if (lane < 16) Lrow[wv][mi] = 1.f / lrow;
  float li[8];
#pragma unroll
  for (int i = 0; i < 8; i++) li[i] = Lrow[wv][i + 8 * hs];

#pragma unroll
  for (int g = 0; g < 4; g++)
#pragma unroll
    for (int i = 0; i < 8; i++) {
      int qr = q0 + i + 8 * hs;
      if (qr < S)
        attn_out[(size_t)(b * S + qr) * 512 + hh * 64 + g * 16 + mi] =
            (bf16)(accO[g][i] * li[i]);
    }
}

// ---------------------------------------------------------------------------
// Router: mean over N of relu projection, then gate softmax
// ---------------------------------------------------------------------------
__global__ __launch_bounds__(256) void rowmean_kernel(const float* __restrict__ R,
                                                      float* __restrict__ r) {
  int i = blockIdx.x * 256 + threadIdx.x;
  if (i >= 4 * 512) return;
  int b = i >> 9, d = i & 511;
  float s = 0.f;
  for (int n = 0; n < 1024; n++) s += R[(size_t)(b * 1024 + n) * 512 + d];
  r[i] = s * (1.f / 1024.f);
}

__global__ __launch_bounds__(32) void gate_kernel(const float* __restrict__ r,
                                                  const float* __restrict__ Wg,
                                                  const float* __restrict__ bg,
                                                  float* __restrict__ gates) {
  __shared__ float lg[16];
  int t = threadIdx.x;
  if (t < 16) {
    int b = t >> 2, e = t & 3;
    float s = bg[e];
    for (int d = 0; d < 512; d++) s += r[b * 512 + d] * Wg[d * 4 + e];
    lg[t] = s;
  }
  __syncthreads();
  if (t < 4) {
    float mx = lg[t * 4];
    for (int e = 1; e < 4; e++) mx = fmaxf(mx, lg[t * 4 + e]);
    float ex[4], s = 0.f;
    for (int e = 0; e < 4; e++) { ex[e] = __expf(lg[t * 4 + e] - mx); s += ex[e]; }
    for (int e = 0; e < 4; e++) gates[t * 4 + e] = ex[e] / s;
  }
}

// ---------------------------------------------------------------------------
// Final LN of CLS row -> latents[e][b][:]
// ---------------------------------------------------------------------------
__global__ __launch_bounds__(128) void lnf_kernel(const float* __restrict__ h,
                                                  const float* __restrict__ g,
                                                  const float* __restrict__ bta,
                                                  float* __restrict__ lat_e) {
  int wv = threadIdx.x >> 5, lane = threadIdx.x & 31;
  const float* x = h + (size_t)wv * 1025 * 512;
  float s = 0.f, sq = 0.f;
#pragma unroll
  for (int j = 0; j < 16; j++) {
    float v = x[lane + j * 32];
    s += v; sq += v * v;
  }
  s = wave_sum(s); sq = wave_sum(sq);
  float mean = s * (1.f / 512.f);
  float var = sq * (1.f / 512.f) - mean * mean;
  float rs = rsqrtf(var + 1e-5f);
#pragma unroll
  for (int j = 0; j < 16; j++) {
    int d = lane + j * 32;
    lat_e[wv * 512 + d] = (x[d] - mean) * rs * g[d] + bta[d];
  }
}

// ---------------------------------------------------------------------------
// Gate-weighted mixing + shared head; writes latent|logits|g_soft to d_out
// ---------------------------------------------------------------------------
__global__ __launch_bounds__(256) void mixhead_kernel(
    const float* __restrict__ gates, const float* __restrict__ latents,
    const float* __restrict__ Wh1, const float* __restrict__ bh1,
    const float* __restrict__ Wh2, const float* __restrict__ bh2,
    float* __restrict__ out) {
  __shared__ float lat[4 * 512];
  __shared__ float hid[4 * 64];
  int t = threadIdx.x;
  for (int i = t; i < 4 * 512; i += 256) {
    int b = i >> 9;
    float v = 0.f;
    for (int e = 0; e < 4; e++) v += gates[b * 4 + e] * latents[(e * 4 + b) * 512 + (i & 511)];
    lat[i] = v;
    out[i] = v;  // latent
  }
  __syncthreads();
  {
    int b = t >> 6, j = t & 63;  // 256 threads = 4*64
    float s = bh1[j];
    for (int d = 0; d < 512; d++) s += lat[b * 512 + d] * Wh1[d * 64 + j];
    hid[t] = fmaxf(s, 0.f);
  }
  __syncthreads();
  if (t < 16) {
    int b = t >> 2, c = t & 3;
    float s = bh2[c];
    for (int j = 0; j < 64; j++) s += hid[b * 64 + j] * Wh2[j * 4 + c];
    out[2048 + t] = s;          // logits
    out[2064 + t] = gates[t];   // g_soft
  }
}

// ---------------------------------------------------------------------------
// Host launcher
// ---------------------------------------------------------------------------
extern "C" void kernel_launch(void* const* d_in, const int* in_sizes, int n_in,
                              void* d_out, int out_size, void* d_ws, size_t ws_size,
                              hipStream_t stream) {
  (void)in_sizes; (void)n_in; (void)out_size; (void)ws_size;
  const float* x     = (const float*)d_in[0];
  const float* Wp    = (const float*)d_in[1];
  const float* bp    = (const float*)d_in[2];
  const float* cls   = (const float*)d_in[3];
  const float* ln1_g = (const float*)d_in[4];
  const float* ln1_b = (const float*)d_in[5];
  const float* Wqkv  = (const float*)d_in[6];
  const float* bqkv  = (const float*)d_in[7];
  const float* Wo    = (const float*)d_in[8];
  const float* bo    = (const float*)d_in[9];
  const float* ln2_g = (const float*)d_in[10];
  const float* ln2_b = (const float*)d_in[11];
  const float* Wm1   = (const float*)d_in[12];
  const float* bm1   = (const float*)d_in[13];
  const float* Wm2   = (const float*)d_in[14];
  const float* bm2   = (const float*)d_in[15];
  const float* lnf_g = (const float*)d_in[16];
  const float* lnf_b = (const float*)d_in[17];
  const float* Wr    = (const float*)d_in[18];
  const float* br    = (const float*)d_in[19];
  const float* Wg    = (const float*)d_in[20];
  const float* bg    = (const float*)d_in[21];
  const float* Wh1   = (const float*)d_in[22];
  const float* bh1   = (const float*)d_in[23];
  const float* Wh2   = (const float*)d_in[24];
  const float* bh2   = (const float*)d_in[25];
  float* out = (float*)d_out;

  const int S = 1025, Spad = 1056, Msk = 4100;  // Msk = B*S

  // --- workspace carve-out ---
  char* wsb = (char*)d_ws;
  size_t off = 0;
  auto take = [&](size_t bytes) -> void* {
    void* p = wsb + off;
    off += (bytes + 255) & ~(size_t)255;
    return p;
  };
  bf16* x_bf    = (bf16*)take((size_t)4 * 1024 * 2048 * 2);
  bf16* Wr_bf   = (bf16*)take((size_t)2048 * 512 * 2);
  bf16* Wp_bf   = (bf16*)take((size_t)4 * 2048 * 512 * 2);
  bf16* Wqkv_bf = (bf16*)take((size_t)4 * 2 * 512 * 1536 * 2);
  bf16* Wo_bf   = (bf16*)take((size_t)4 * 2 * 512 * 512 * 2);
  bf16* Wm1_bf  = (bf16*)take((size_t)4 * 2 * 512 * 512 * 2);
  bf16* Wm2_bf  = (bf16*)take((size_t)4 * 2 * 512 * 512 * 2);
  float* h      = (float*)take((size_t)4 * 1025 * 512 * 4);
  bf16* hn_bf   = (bf16*)take((size_t)Msk * 512 * 2);
  bf16* qkv_bf  = (bf16*)take((size_t)Msk * 1536 * 2);
  float* Rfull  = (float*)qkv_bf;  // 4096*512*4 B <= qkv buffer; used before it
  bf16* kT      = (bf16*)take((size_t)4 * 8 * 64 * Spad * 2);
  bf16* attn_bf = (bf16*)take((size_t)Msk * 512 * 2);
  bf16* g_bf    = (bf16*)take((size_t)Msk * 512 * 2);
  float* rbuf   = (float*)take(2048 * 4);
  float* gates  = (float*)take(64);
  float* lat    = (float*)take(4 * 4 * 512 * 4);

  auto cvt = [&](const float* in, bf16* o, long n) {
    int blocks = (int)((n + 255) / 256);
    cvt_kernel<<<blocks, 256, 0, stream>>>(in, o, n);
  };

  // --- bf16 conversion of activations and weights ---
  cvt(x, x_bf, (long)4 * 1024 * 2048);
  cvt(Wr, Wr_bf, (long)2048 * 512);
  cvt(Wp, Wp_bf, (long)4 * 2048 * 512);
  cvt(Wqkv, Wqkv_bf, (long)4 * 2 * 512 * 1536);
  cvt(Wo, Wo_bf, (long)4 * 2 * 512 * 512);
  cvt(Wm1, Wm1_bf, (long)4 * 2 * 512 * 512);
  cvt(Wm2, Wm2_bf, (long)4 * 2 * 512 * 512);

  // --- router: relu(x@Wr+br) -> mean over N -> softmax gate ---
  gemm_kernel<ACT_RELU, false, false, false>
      <<<dim3(512 / 64, (4096 + 127) / 128), 256, 0, stream>>>(
          x_bf, Wr_bf, br, Rfull, nullptr, nullptr, 4096, 512, 2048, 1);
  rowmean_kernel<<<8, 256, 0, stream>>>(Rfull, rbuf);
  gate_kernel<<<1, 32, 0, stream>>>(rbuf, Wg, bg, gates);

  // --- experts ---
  for (int e = 0; e < 4; e++) {
    // patch projection into h[b, 1.., :] (row remap), then CLS fill
    gemm_kernel<ACT_NONE, false, false, true>
        <<<dim3(512 / 64, (4096 + 127) / 128), 256, 0, stream>>>(
            x_bf, Wp_bf + (size_t)e * 2048 * 512, bp + e * 512, h, nullptr,
            nullptr, 4096, 512, 2048, 1024);
    clsfill_kernel<<<8, 256, 0, stream>>>(h, cls + e * 512);

    for (int l = 0; l < 2; l++) {
      const int el = e * 2 + l;
      // LN1 -> QKV
      ln_kernel<<<(Msk + 7) / 8, 256, 0, stream>>>(
          h, ln1_g + el * 512, ln1_b + el * 512, hn_bf, Msk);
      gemm_kernel<ACT_NONE, false, true, false>
          <<<dim3(1536 / 64, (Msk + 127) / 128), 256, 0, stream>>>(
              hn_bf, Wqkv_bf + (size_t)el * 512 * 1536, bqkv + el * 1536,
              nullptr, qkv_bf, nullptr, Msk, 1536, 512, 1);
      // K transpose + flash attention
      kt_kernel<<<(4 * 8 * 64 * Spad + 255) / 256, 256, 0, stream>>>(
          qkv_bf, kT, S, Spad);
      attn_kernel<<<(4 * 8 * ((S + 15) / 16) + 7) / 8, 256, 0, stream>>>(
          qkv_bf, kT, attn_bf, S, Spad);
      // h += attn @ Wo + bo
      gemm_kernel<ACT_NONE, true, false, false>
          <<<dim3(512 / 64, (Msk + 127) / 128), 256, 0, stream>>>(
              attn_bf, Wo_bf + (size_t)el * 512 * 512, bo + el * 512, h,
              nullptr, h, Msk, 512, 512, 1);
      // LN2 -> MLP
      ln_kernel<<<(Msk + 7) / 8, 256, 0, stream>>>(
          h, ln2_g + el * 512, ln2_b + el * 512, hn_bf, Msk);
      gemm_kernel<ACT_GELU, false, true, false>
          <<<dim3(512 / 64, (Msk + 127) / 128), 256, 0, stream>>>(
              hn_bf, Wm1_bf + (size_t)el * 512 * 512, bm1 + el * 512, nullptr,
              g_bf, nullptr, Msk, 512, 512, 1);
      gemm_kernel<ACT_NONE, true, false, false>
          <<<dim3(512 / 64, (Msk + 127) / 128), 256, 0, stream>>>(
              g_bf, Wm2_bf + (size_t)el * 512 * 512, bm2 + el * 512, h,
              nullptr, h, Msk, 512, 512, 1);
    }
    // final LN of CLS row
    lnf_kernel<<<1, 128, 0, stream>>>(h, lnf_g + e * 512, lnf_b + e * 512,
                                      lat + (size_t)e * 4 * 512);
  }

  // --- mix + head ---
  mixhead_kernel<<<1, 256, 0, stream>>>(gates, lat, Wh1, bh1, Wh2, bh2, out);
}
